// DQTLinear_15058155339855
// MI455X (gfx1250) — matrix operand order repaired and checked by reading, not compiled, not run on hardware
//
#include <hip/hip_runtime.h>

// ---- CDNA5 (gfx1250) ternary-quantized linear:  out = x @ (ternary*scales)^T
// M=8192 (2x4096), N=4096 (OUT), K=4096 (IN), group size 128 along K.
// Fast path: pre-dequantize to bf16 in d_ws (scale baked into weights), then
// pure bf16 WMMA GEMM; LDS tiles filled with GLOBAL_LOAD_ASYNC_TO_LDS_B128.

typedef __attribute__((ext_vector_type(16))) __bf16        v16bf;
typedef __attribute__((ext_vector_type(8)))  __bf16        v8bf;
typedef __attribute__((ext_vector_type(8)))  float         v8f;
typedef __attribute__((ext_vector_type(4)))  float         f32x4;
typedef __attribute__((ext_vector_type(4)))  unsigned int  u32x4;

#define K_DIM   4096
#define N_DIM   4096
#define M_DIM   8192
#define GSZ     128
#define NGROUP  (K_DIM / GSZ)   // 32 groups per weight row

// ======================= pre-pass 1: x fp32 -> bf16 ==========================
__global__ __launch_bounds__(256) void convert_x_bf16(
    const float* __restrict__ x, unsigned short* __restrict__ xbf)
{
    const size_t i = ((size_t)blockIdx.x * 256 + threadIdx.x) * 8;
    const f32x4 f0 = __builtin_nontemporal_load(reinterpret_cast<const f32x4*>(x + i));
    const f32x4 f1 = __builtin_nontemporal_load(reinterpret_cast<const f32x4*>(x + i + 4));
    const float fv[8] = {f0.x, f0.y, f0.z, f0.w, f1.x, f1.y, f1.z, f1.w};
    u32x4 o;
    unsigned int p[4];
    #pragma unroll
    for (int q = 0; q < 4; ++q) {
        const unsigned int lo = __builtin_bit_cast(unsigned short, (__bf16)fv[2 * q]);
        const unsigned int hi = __builtin_bit_cast(unsigned short, (__bf16)fv[2 * q + 1]);
        p[q] = lo | (hi << 16);
    }
    o.x = p[0]; o.y = p[1]; o.z = p[2]; o.w = p[3];
    *reinterpret_cast<u32x4*>(xbf + i) = o;
}

// ============ pre-pass 2: w = bf16(ternary * group_scale) ====================
__global__ __launch_bounds__(256) void convert_w_bf16(
    const signed char* __restrict__ w, const float* __restrict__ scales,
    unsigned short* __restrict__ wbf)
{
    const size_t e0 = ((size_t)blockIdx.x * 256 + threadIdx.x) * 16;
    const u32x4 raw = __builtin_nontemporal_load(reinterpret_cast<const u32x4*>(w + e0));
    const float sc = scales[e0 >> 7];           // flat group index = flat_elem / 128
    const unsigned int d[4] = {raw.x, raw.y, raw.z, raw.w};
    unsigned int ow[8];
    #pragma unroll
    for (int q = 0; q < 4; ++q) {
        unsigned int pk[2];
        #pragma unroll
        for (int hfs = 0; hfs < 2; ++hfs) {
            const signed char s0 = (signed char)(d[q] >> (16 * hfs));
            const signed char s1 = (signed char)(d[q] >> (16 * hfs + 8));
            const unsigned int lo = __builtin_bit_cast(unsigned short, (__bf16)((float)s0 * sc));
            const unsigned int hi = __builtin_bit_cast(unsigned short, (__bf16)((float)s1 * sc));
            pk[hfs] = lo | (hi << 16);
        }
        ow[2 * q] = pk[0]; ow[2 * q + 1] = pk[1];
    }
    u32x4 s0v, s1v;
    s0v.x = ow[0]; s0v.y = ow[1]; s0v.z = ow[2]; s0v.w = ow[3];
    s1v.x = ow[4]; s1v.y = ow[5]; s1v.z = ow[6]; s1v.w = ow[7];
    *reinterpret_cast<u32x4*>(wbf + e0)     = s0v;
    *reinterpret_cast<u32x4*>(wbf + e0 + 8) = s1v;
}

// ======================= main bf16 WMMA GEMM ================================
// Block: 256 threads (8 waves). Block tile 128(M) x 128(N).
// Wave tile 64(M) x 32(N) = 4x2 WMMA tiles. K-block 64 staged in LDS.
// LDS row stride padded to 72 halves (36 dwords) -> conflict-free ds_load_b128.
// LDS tiles filled via async global->LDS DMA (ASYNCcnt), no VGPR staging.
#define BLK_M   128
#define BLK_N   128
#define BLK_K   64
#define LDSTR   72     // padded stride in halves

__global__ __launch_bounds__(256) void gemm_bf16_wmma(
    const unsigned short* __restrict__ xbf,   // [M, K] bf16 bits
    const unsigned short* __restrict__ wbf,   // [N, K] bf16 bits (scale baked in)
    float* __restrict__ out)                  // [M, N] fp32
{
    __shared__ unsigned short a_lds[BLK_M * LDSTR];  // 18.4 KB
    __shared__ unsigned short b_lds[BLK_N * LDSTR];  // 18.4 KB

    const int tid  = threadIdx.x;
    const int lane = tid & 31;
    const int wid  = tid >> 5;     // 0..7
    const int wm   = wid >> 2;     // 0..1 : wave row (64 M-rows each)
    const int wn   = wid & 3;      // 0..3 : wave col (32 N-cols each)

    const int half = lane >> 4;
    const int l16  = lane & 15;

    const int block_n0 = blockIdx.x * BLK_N;
    const int block_m0 = blockIdx.y * BLK_M;

    // LDS byte offsets for async DMA destinations (flat addr low bits = LDS offset)
    const unsigned int a_base = (unsigned int)(unsigned long long)(&a_lds[0]);
    const unsigned int b_base = (unsigned int)(unsigned long long)(&b_lds[0]);

    // this thread's fixed chunk coordinates for the fill phase
    const int fr = tid >> 3;      // 0..31 base row (advances by 32 per it)
    const int fc = tid & 7;       // 8-half chunk within K-block

    v8f acc[4][2];
    #pragma unroll
    for (int mi = 0; mi < 4; ++mi)
        #pragma unroll
        for (int ni = 0; ni < 2; ++ni)
            acc[mi][ni] = v8f{0.f, 0.f, 0.f, 0.f, 0.f, 0.f, 0.f, 0.f};

    for (int k0 = 0; k0 < K_DIM; k0 += BLK_K) {
        __syncthreads();  // previous tile fully consumed

        // ---- async fill A: 128 rows x 64 k (1024 16B chunks; 4 per thread) ----
        #pragma unroll
        for (int it = 0; it < 4; ++it) {
            const int r = fr + it * 32;
            const unsigned short* gsrc =
                &xbf[(size_t)(block_m0 + r) * K_DIM + (size_t)(k0 + fc * 8)];
            const unsigned int ldst = a_base + (unsigned int)((r * LDSTR + fc * 8) * 2);
            asm volatile("global_load_async_to_lds_b128 %0, %1, off"
                         :: "v"(ldst), "v"(gsrc) : "memory");
        }
        // ---- async fill B: 128 cols x 64 k ----
        #pragma unroll
        for (int it = 0; it < 4; ++it) {
            const int r = fr + it * 32;
            const unsigned short* gsrc =
                &wbf[(size_t)(block_n0 + r) * K_DIM + (size_t)(k0 + fc * 8)];
            const unsigned int ldst = b_base + (unsigned int)((r * LDSTR + fc * 8) * 2);
            asm volatile("global_load_async_to_lds_b128 %0, %1, off"
                         :: "v"(ldst), "v"(gsrc) : "memory");
        }
        asm volatile("s_wait_asynccnt 0" ::: "memory");
        __syncthreads();  // all waves' DMA visible

        // ---- 2 WMMA K-steps (K=32 each), 8 WMMAs per step ----
        #pragma unroll
        for (int t = 0; t < 2; ++t) {
            v16bf afrag[4];
            #pragma unroll
            for (int mi = 0; mi < 4; ++mi) {
                const int row = wm * 64 + mi * 16 + l16;
                const int kb  = t * 32 + half * 16;
                const v8bf lo = *reinterpret_cast<const v8bf*>(&a_lds[row * LDSTR + kb]);
                const v8bf hi = *reinterpret_cast<const v8bf*>(&a_lds[row * LDSTR + kb + 8]);
                afrag[mi] = __builtin_shufflevector(lo, hi,
                    0, 1, 2, 3, 4, 5, 6, 7, 8, 9, 10, 11, 12, 13, 14, 15);
            }
            v16bf bfrag[2];
            #pragma unroll
            for (int ni = 0; ni < 2; ++ni) {
                const int col = wn * 32 + ni * 16 + l16;
                const int kb  = t * 32 + half * 16;
                const v8bf lo = *reinterpret_cast<const v8bf*>(&b_lds[col * LDSTR + kb]);
                const v8bf hi = *reinterpret_cast<const v8bf*>(&b_lds[col * LDSTR + kb + 8]);
                bfrag[ni] = __builtin_shufflevector(lo, hi,
                    0, 1, 2, 3, 4, 5, 6, 7, 8, 9, 10, 11, 12, 13, 14, 15);
            }
            #pragma unroll
            for (int mi = 0; mi < 4; ++mi)
                #pragma unroll
                for (int ni = 0; ni < 2; ++ni)
                    acc[mi][ni] = __builtin_amdgcn_wmma_f32_16x16x32_bf16(
                        false, afrag[mi], false, bfrag[ni],
                        (short)0, acc[mi][ni], false, false);
        }
    }

    // ---- write out ----
    #pragma unroll
    for (int mi = 0; mi < 4; ++mi) {
        const int row0 = block_m0 + wm * 64 + mi * 16 + half * 8;
        #pragma unroll
        for (int ni = 0; ni < 2; ++ni) {
            const int col = block_n0 + wn * 32 + ni * 16 + l16;
            #pragma unroll
            for (int r = 0; r < 8; ++r)
                __builtin_nontemporal_store(acc[mi][ni][r],
                    &out[(size_t)(row0 + r) * N_DIM + col]);
        }
    }
}

// =============== fallback (ws too small): round-1 fused kernel ===============
#define FBLK_M   128
#define FBLK_N   64

__global__ __launch_bounds__(256) void ternary_gemm_wmma(
    const float* __restrict__ x, const signed char* __restrict__ w,
    const float* __restrict__ scales, float* __restrict__ out)
{
    __shared__ unsigned short a_lds[FBLK_M * GSZ];
    __shared__ unsigned short b_lds[FBLK_N * GSZ];

    const int tid  = threadIdx.x;
    const int lane = tid & 31;
    const int wid  = tid >> 5;
    const int wm   = wid & 3;
    const int wn   = wid >> 2;
    const int half = lane >> 4;
    const int l16  = lane & 15;
    const int block_n0 = blockIdx.x * FBLK_N;
    const int block_m0 = blockIdx.y * FBLK_M;

    v8f acc[2][2];
    #pragma unroll
    for (int mi = 0; mi < 2; ++mi)
        #pragma unroll
        for (int ni = 0; ni < 2; ++ni)
            acc[mi][ni] = v8f{0.f, 0.f, 0.f, 0.f, 0.f, 0.f, 0.f, 0.f};

    for (int g = 0; g < NGROUP; ++g) {
        const int k0 = g * GSZ;
        __syncthreads();
        #pragma unroll
        for (int it = 0; it < 16; ++it) {
            const int idx = tid + it * 256;
            const int r   = idx >> 5;
            const int c4  = idx & 31;
            const f32x4 f = *reinterpret_cast<const f32x4*>(
                &x[(size_t)(block_m0 + r) * K_DIM + (size_t)(k0 + c4 * 4)]);
            const unsigned int u0 = __builtin_bit_cast(unsigned short, (__bf16)f.x);
            const unsigned int u1 = __builtin_bit_cast(unsigned short, (__bf16)f.y);
            const unsigned int u2 = __builtin_bit_cast(unsigned short, (__bf16)f.z);
            const unsigned int u3 = __builtin_bit_cast(unsigned short, (__bf16)f.w);
            uint2 p; p.x = u0 | (u1 << 16); p.y = u2 | (u3 << 16);
            *reinterpret_cast<uint2*>(&a_lds[r * GSZ + c4 * 4]) = p;
        }
        #pragma unroll
        for (int it = 0; it < 2; ++it) {
            const int idx = tid + it * 256;
            const int col = idx >> 3;
            const int c16 = idx & 7;
            const u32x4 raw = *reinterpret_cast<const u32x4*>(
                &w[(size_t)(block_n0 + col) * K_DIM + (size_t)(k0 + c16 * 16)]);
            const unsigned int d[4] = {raw.x, raw.y, raw.z, raw.w};
            unsigned int ow[8];
            #pragma unroll
            for (int q = 0; q < 4; ++q) {
                const unsigned int v  = d[q];
                const unsigned int b0 =  v        & 0xFFu;
                const unsigned int b1 = (v >> 8)  & 0xFFu;
                const unsigned int b2 = (v >> 16) & 0xFFu;
                const unsigned int b3 =  v >> 24;
                unsigned int lo  = (b0 ? 0x3F80u : 0u) | ((b0 & 0x80u) << 8);
                lo |= ((b1 ? 0x3F80u : 0u) | ((b1 & 0x80u) << 8)) << 16;
                unsigned int hi  = (b2 ? 0x3F80u : 0u) | ((b2 & 0x80u) << 8);
                hi |= ((b3 ? 0x3F80u : 0u) | ((b3 & 0x80u) << 8)) << 16;
                ow[2 * q] = lo; ow[2 * q + 1] = hi;
            }
            u32x4 s0, s1;
            s0.x = ow[0]; s0.y = ow[1]; s0.z = ow[2]; s0.w = ow[3];
            s1.x = ow[4]; s1.y = ow[5]; s1.z = ow[6]; s1.w = ow[7];
            *reinterpret_cast<u32x4*>(&b_lds[col * GSZ + c16 * 16])     = s0;
            *reinterpret_cast<u32x4*>(&b_lds[col * GSZ + c16 * 16 + 8]) = s1;
        }
        __syncthreads();

        v8f gacc[2][2];
        #pragma unroll
        for (int mi = 0; mi < 2; ++mi)
            #pragma unroll
            for (int ni = 0; ni < 2; ++ni)
                gacc[mi][ni] = v8f{0.f, 0.f, 0.f, 0.f, 0.f, 0.f, 0.f, 0.f};

        #pragma unroll
        for (int t = 0; t < 4; ++t) {
            v16bf afrag[2];
            #pragma unroll
            for (int mi = 0; mi < 2; ++mi) {
                const int row = wm * 32 + mi * 16 + l16;
                const int kb  = t * 32 + half * 16;
                const v8bf lo = *reinterpret_cast<const v8bf*>(&a_lds[row * GSZ + kb]);
                const v8bf hi = *reinterpret_cast<const v8bf*>(&a_lds[row * GSZ + kb + 8]);
                afrag[mi] = __builtin_shufflevector(lo, hi,
                    0, 1, 2, 3, 4, 5, 6, 7, 8, 9, 10, 11, 12, 13, 14, 15);
            }
            v16bf bfrag[2];
            #pragma unroll
            for (int ni = 0; ni < 2; ++ni) {
                const int col = wn * 32 + ni * 16 + l16;
                const int kb  = t * 32 + half * 16;
                bfrag[ni] = *reinterpret_cast<const v16bf*>(&b_lds[col * GSZ + kb]);
            }
            #pragma unroll
            for (int mi = 0; mi < 2; ++mi)
                #pragma unroll
                for (int ni = 0; ni < 2; ++ni)
                    gacc[mi][ni] = __builtin_amdgcn_wmma_f32_16x16x32_bf16(
                        false, afrag[mi], false, bfrag[ni],
                        (short)0, gacc[mi][ni], false, false);
        }
        #pragma unroll
        for (int ni = 0; ni < 2; ++ni) {
            const int col = block_n0 + wn * 32 + ni * 16 + l16;
            const float sc = scales[(size_t)col * NGROUP + g];
            #pragma unroll
            for (int mi = 0; mi < 2; ++mi)
                #pragma unroll
                for (int r = 0; r < 8; ++r)
                    acc[mi][ni][r] += sc * gacc[mi][ni][r];
        }
    }
    #pragma unroll
    for (int mi = 0; mi < 2; ++mi) {
        const int row0 = block_m0 + wm * 32 + mi * 16 + half * 8;
        #pragma unroll
        for (int ni = 0; ni < 2; ++ni) {
            const int col = block_n0 + wn * 32 + ni * 16 + l16;
            #pragma unroll
            for (int r = 0; r < 8; ++r)
                __builtin_nontemporal_store(acc[mi][ni][r],
                    &out[(size_t)(row0 + r) * N_DIM + col]);
        }
    }
}

extern "C" void kernel_launch(void* const* d_in, const int* in_sizes, int n_in,
                              void* d_out, int out_size, void* d_ws, size_t ws_size,
                              hipStream_t stream) {
    (void)in_sizes; (void)n_in; (void)out_size;
    const float*       x      = (const float*)d_in[0];        // (2,4096,4096) fp32
    const signed char* tern   = (const signed char*)d_in[1];  // (4096,4096) int8
    const float*       scales = (const float*)d_in[2];        // (131072,) fp32
    float*             out    = (float*)d_out;                // (2,4096,4096) fp32

    const size_t x_elems = (size_t)M_DIM * K_DIM;             // 33,554,432
    const size_t w_elems = (size_t)N_DIM * K_DIM;             // 16,777,216
    const size_t need    = x_elems * 2 + w_elems * 2;         // ~100.7 MB

    if (ws_size >= need) {
        unsigned short* xbf = (unsigned short*)d_ws;
        unsigned short* wbf = xbf + x_elems;
        convert_x_bf16<<<dim3((unsigned)(x_elems / (256 * 8))), dim3(256), 0, stream>>>(x, xbf);
        convert_w_bf16<<<dim3((unsigned)(w_elems / (256 * 16))), dim3(256), 0, stream>>>(tern, scales, wbf);
        gemm_bf16_wmma<<<dim3(N_DIM / BLK_N, M_DIM / BLK_M), dim3(256), 0, stream>>>(xbf, wbf, out);
    } else {
        ternary_gemm_wmma<<<dim3(N_DIM / FBLK_N, M_DIM / FBLK_M), dim3(256), 0, stream>>>(
            x, tern, scales, out);
    }
}
